// MultiHeadAttention_78847009620307
// MI455X (gfx1250) — matrix operand order repaired and checked by reading, not compiled, not run on hardware
//
#include <hip/hip_runtime.h>

// ---------------------------------------------------------------------------
// MI455X (gfx1250) multi-head causal attention forward, bf16 WMMA pipeline.
// GLOBAL_LOAD_ASYNC_TO_LDS_B128 staging, double-buffered LDS everywhere,
// pre-transposed V, DPP-based softmax reductions (LDS pipe left to WMMA).
// ---------------------------------------------------------------------------

typedef __attribute__((ext_vector_type(16))) __bf16 v16bf;
typedef __attribute__((ext_vector_type(8)))  __bf16 v8bf;
typedef __attribute__((ext_vector_type(8)))  float  v8f;

#define D_MODEL 2048
#define NUM_HEAD 16
#define HEAD_DIM 128
#define SEQ 2048
#define BATCH 2
#define MTOT (BATCH * SEQ) /* 4096 */

#if defined(__has_builtin)
#if __has_builtin(__builtin_amdgcn_global_load_async_to_lds_b128)
#define USE_ASYNC_LDS 1
#endif
#if __has_builtin(__builtin_amdgcn_update_dpp)
#define HAVE_DPP 1
#endif
#endif
#ifndef USE_ASYNC_LDS
#define USE_ASYNC_LDS 0
#endif
#ifndef HAVE_DPP
#define HAVE_DPP 0
#endif

#if USE_ASYNC_LDS
// Builtin signature (probe-confirmed): (AS1 v4i*, AS3 v4i*, imm offset, cpol).
typedef int v4i_t __attribute__((vector_size(16)));
typedef v4i_t __attribute__((address_space(1))) g_v4i;
typedef v4i_t __attribute__((address_space(3))) s_v4i;
static __device__ __forceinline__ void async_cp16(void* lds_dst, const void* gsrc) {
  __builtin_amdgcn_global_load_async_to_lds_b128(
      (g_v4i*)(uintptr_t)gsrc, (s_v4i*)(uintptr_t)lds_dst, 0, 0);
}
#endif

static __device__ __forceinline__ void wait_async0() {
#if defined(__has_builtin) && __has_builtin(__builtin_amdgcn_s_wait_asynccnt)
  __builtin_amdgcn_s_wait_asynccnt(0);
#else
  asm volatile("s_wait_asynccnt 0" ::: "memory");
#endif
}

// ---- 16-lane butterfly reductions on the VALU DPP path (no LDS traffic) ----
#if HAVE_DPP
template <int CTRL>
static __device__ __forceinline__ float dpp_perm_f(float v) {
  int i = __builtin_bit_cast(int, v);
  int r = __builtin_amdgcn_update_dpp(i, i, CTRL, 0xf, 0xf, true);
  return __builtin_bit_cast(float, r);
}
static __device__ __forceinline__ float xlane16_max(float v) {
  v = fmaxf(v, dpp_perm_f<0xB1>(v));  // quad_perm [1,0,3,2]  (xor 1)
  v = fmaxf(v, dpp_perm_f<0x4E>(v));  // quad_perm [2,3,0,1]  (xor 2)
  v = fmaxf(v, dpp_perm_f<0x141>(v)); // ROW_HALF_MIRROR      (xor 7)
  v = fmaxf(v, dpp_perm_f<0x140>(v)); // ROW_MIRROR           (xor 15)
  return v;
}
static __device__ __forceinline__ float xlane16_sum(float v) {
  v += dpp_perm_f<0xB1>(v);
  v += dpp_perm_f<0x4E>(v);
  v += dpp_perm_f<0x141>(v);
  v += dpp_perm_f<0x140>(v);
  return v;
}
#else
static __device__ __forceinline__ float xlane16_max(float v) {
#pragma unroll
  for (int off = 8; off; off >>= 1) v = fmaxf(v, __shfl_xor(v, off, 16));
  return v;
}
static __device__ __forceinline__ float xlane16_sum(float v) {
#pragma unroll
  for (int off = 8; off; off >>= 1) v += __shfl_xor(v, off, 16);
  return v;
}
#endif

static __device__ __forceinline__ __bf16 f2bf(float f) {
  unsigned u = __builtin_bit_cast(unsigned, f);
  u += 0x7fffu + ((u >> 16) & 1u);           // round-to-nearest-even
  unsigned short s = (unsigned short)(u >> 16);
  return __builtin_bit_cast(__bf16, s);
}

static __device__ __forceinline__ v16bf cat8(v8bf lo, v8bf hi) {
  return __builtin_shufflevector(lo, hi, 0, 1, 2, 3, 4, 5, 6, 7,
                                 8, 9, 10, 11, 12, 13, 14, 15);
}

static __device__ __forceinline__ v8f zero8() {
  v8f z;
#pragma unroll
  for (int i = 0; i < 8; ++i) z[i] = 0.0f;
  return z;
}

// ---------------------------------------------------------------------------
// fp32 -> bf16 conversion, 4 elements per thread
// ---------------------------------------------------------------------------
__global__ __launch_bounds__(256) void cvt_f32_bf16(
    const float* __restrict__ in, __bf16* __restrict__ out, int n4) {
  int i = blockIdx.x * blockDim.x + threadIdx.x;
  if (i < n4) {
    float4 v = ((const float4*)in)[i];
    ushort4 o;
    o.x = __builtin_bit_cast(unsigned short, f2bf(v.x));
    o.y = __builtin_bit_cast(unsigned short, f2bf(v.y));
    o.z = __builtin_bit_cast(unsigned short, f2bf(v.z));
    o.w = __builtin_bit_cast(unsigned short, f2bf(v.w));
    ((ushort4*)out)[i] = o;
  }
}

// ---------------------------------------------------------------------------
// One-time V transpose: Vt_g[bh][d][t] = V[b*SEQ + t][h*HEAD_DIM + d].
// ---------------------------------------------------------------------------
__global__ __launch_bounds__(256) void transpose_v(
    const __bf16* __restrict__ Vb, __bf16* __restrict__ Vt_g) {
  __shared__ alignas(64) __bf16 tile[64][80]; // padded rows (160B, 16B-aligned)
  const int tid = threadIdx.x;
  const int t0 = blockIdx.x * 64;
  const int d0 = blockIdx.y * 64;
  const int bh = blockIdx.z;
  const int b = bh >> 4;
  const int h = bh & 15;

#pragma unroll
  for (int rep = 0; rep < 2; ++rep) {
    const int c = tid + rep * 256;   // 0..511
    const int tr = c >> 3;           // 0..63
    const int dc = (c & 7) << 3;     // 0..56
    *(v8bf*)&tile[tr][dc] =
        *(const v8bf*)&Vb[(size_t)(b * SEQ + t0 + tr) * D_MODEL + h * HEAD_DIM + d0 + dc];
  }
  __syncthreads();
#pragma unroll
  for (int rep = 0; rep < 2; ++rep) {
    const int c = tid + rep * 256;
    const int dr = c >> 3;           // 0..63
    const int tc = (c & 7) << 3;     // 0..56
    v8bf o;
#pragma unroll
    for (int e = 0; e < 8; ++e) o[e] = tile[tc + e][dr];
    *(v8bf*)&Vt_g[((size_t)bh * HEAD_DIM + d0 + dr) * SEQ + t0 + tc] = o;
  }
}

// ---------------------------------------------------------------------------
// C[M,N] = outScale * (A[M,K] @ W[N,K]^T)  (bf16 in, fp32 accum, bf16/f32 out)
// Block tile 128x256, 8 waves (2 M x 4 N), wave tile 64x64 -> 16 WMMA tiles.
// ---------------------------------------------------------------------------
template <bool OUT_F32>
__global__ __launch_bounds__(256) void gemm_bf16t(
    const __bf16* __restrict__ A, const __bf16* __restrict__ W,
    void* __restrict__ Cout, int M, int N, int Kd, float outScale) {
  __shared__ alignas(64) __bf16 As[2][128][32];
  __shared__ alignas(64) __bf16 Ws[2][256][32];

  const int tid = threadIdx.x;
  const int lane = tid & 31;
  const int wid = tid >> 5;
  const int wm = wid & 1;   // 0..1  (M waves)
  const int wn = wid >> 1;  // 0..3  (N waves)
  const int g = lane >> 4;  // lane-half
  const int lm = lane & 15;

  const int rowBase = blockIdx.y * 128;
  const int colBase = blockIdx.x * 256;

  v8f acc[4][4];
#pragma unroll
  for (int mi = 0; mi < 4; ++mi)
#pragma unroll
    for (int ni = 0; ni < 4; ++ni) acc[mi][ni] = zero8();

  auto stage = [&](int buf, int k0) {
#pragma unroll
    for (int rep = 0; rep < 2; ++rep) {
      const int idx = tid + rep * 256;   // 0..511
      const int r = idx >> 2;            // 0..127
      const int c = (idx & 3) << 3;      // 0,8,16,24
#if USE_ASYNC_LDS
      async_cp16(&As[buf][r][c], &A[(size_t)(rowBase + r) * Kd + k0 + c]);
#else
      *(uint4*)&As[buf][r][c] = *(const uint4*)&A[(size_t)(rowBase + r) * Kd + k0 + c];
#endif
    }
#pragma unroll
    for (int rep = 0; rep < 4; ++rep) {
      const int idx = tid + rep * 256;   // 0..1023
      const int r = idx >> 2;            // 0..255
      const int c = (idx & 3) << 3;
#if USE_ASYNC_LDS
      async_cp16(&Ws[buf][r][c], &W[(size_t)(colBase + r) * Kd + k0 + c]);
#else
      *(uint4*)&Ws[buf][r][c] = *(const uint4*)&W[(size_t)(colBase + r) * Kd + k0 + c];
#endif
    }
  };

  const int nK = Kd >> 5;
  stage(0, 0);
  for (int kt = 0; kt < nK; ++kt) {
    const int cur = kt & 1;
#if USE_ASYNC_LDS
    wait_async0();          // my async slab writes have landed in LDS
    __syncthreads();        // everyone's slab complete; prev compute done
    if (kt + 1 < nK) stage(cur ^ 1, (kt + 1) << 5);
#else
    __syncthreads();
    if (kt + 1 < nK) stage(cur ^ 1, (kt + 1) << 5);
    __syncthreads();
#endif

    v16bf af[4], bfr[4];
#pragma unroll
    for (int mi = 0; mi < 4; ++mi) {
      const int ar = wm * 64 + mi * 16 + lm;
      // A fragment: half g holds K = {16g..16g+15} (matches B)
      af[mi] = cat8(*(const v8bf*)&As[cur][ar][16 * g],
                    *(const v8bf*)&As[cur][ar][16 * g + 8]);
    }
#pragma unroll
    for (int ni = 0; ni < 4; ++ni) {
      const int br = wn * 64 + ni * 16 + lm;
      bfr[ni] = *(const v16bf*)&Ws[cur][br][16 * g];
    }
#pragma unroll
    for (int mi = 0; mi < 4; ++mi)
#pragma unroll
      for (int ni = 0; ni < 4; ++ni)
        acc[mi][ni] = __builtin_amdgcn_wmma_f32_16x16x32_bf16(
            false, af[mi], false, bfr[ni], (short)0, acc[mi][ni], false, false);
  }

#pragma unroll
  for (int mi = 0; mi < 4; ++mi)
#pragma unroll
    for (int ni = 0; ni < 4; ++ni)
#pragma unroll
      for (int r = 0; r < 8; ++r) {
        const int gm = rowBase + wm * 64 + mi * 16 + g * 8 + r;
        const int gn = colBase + wn * 64 + ni * 16 + lm;
        const float v = acc[mi][ni][r] * outScale;
        if constexpr (OUT_F32)
          ((float*)Cout)[(size_t)gm * N + gn] = v;
        else
          ((__bf16*)Cout)[(size_t)gm * N + gn] = f2bf(v);
      }
}

// ---------------------------------------------------------------------------
// Flash attention forward (causal). Grid: (SEQ/128, BATCH*NUM_HEAD).
// 8 waves; wave w owns query rows [qbase+16w, +16). Key/value tiles (32 keys)
// double-buffered in LDS via async loads; online softmax in fp32 (Q is
// pre-scaled by 1/sqrt(dk) in the Q-projection epilogue).
// ---------------------------------------------------------------------------
__global__ __launch_bounds__(256) void attn_fwd(
    const __bf16* __restrict__ Q, const __bf16* __restrict__ Km,
    const __bf16* __restrict__ Vt_g, __bf16* __restrict__ O) {
  __shared__ alignas(64) __bf16 Ks[2][32][128];   // keys, row-major
  __shared__ alignas(64) __bf16 Vt[2][128][32];   // values, dk-major
  __shared__ alignas(64) __bf16 Ps[8][16][32];    // per-wave P scratch

  const int tid = threadIdx.x;
  const int lane = tid & 31;
  const int w = tid >> 5;
  const int g = lane >> 4;
  const int lm = lane & 15;

  const int bh = blockIdx.y;
  const int b = bh >> 4;
  const int qbase = blockIdx.x * 128;
  const size_t rowOff = (size_t)b * SEQ;
  const int colOff = (bh & 15) * HEAD_DIM;

  auto stage = [&](int buf, int j0) {
#pragma unroll
    for (int rep = 0; rep < 2; ++rep) {
      const int idx = tid + rep * 256;  // 0..511
      { // K: 32 rows x 128 cols
        const int kr = idx >> 4;        // 0..31
        const int c = (idx & 15) << 3;  // 0..120
        const size_t gsrc = (rowOff + j0 + kr) * D_MODEL + colOff + c;
#if USE_ASYNC_LDS
        async_cp16(&Ks[buf][kr][c], &Km[gsrc]);
#else
        *(uint4*)&Ks[buf][kr][c] = *(const uint4*)&Km[gsrc];
#endif
      }
      { // Vt: 128 dk-rows x 32 keys
        const int dr = idx >> 2;        // 0..127
        const int tc = (idx & 3) << 3;  // 0,8,16,24
        const size_t gsrc = ((size_t)bh * HEAD_DIM + dr) * SEQ + j0 + tc;
#if USE_ASYNC_LDS
        async_cp16(&Vt[buf][dr][tc], &Vt_g[gsrc]);
#else
        *(uint4*)&Vt[buf][dr][tc] = *(const uint4*)&Vt_g[gsrc];
#endif
      }
    }
  };

  // Load the wave's 16x128 Q tile as 4 A-fragments (K-depth 32 each).
  v16bf aq[4];
  {
    const size_t qrow = rowOff + qbase + w * 16 + lm;
    const __bf16* qp = Q + qrow * D_MODEL + colOff;
#pragma unroll
    for (int kk = 0; kk < 4; ++kk)
      aq[kk] = cat8(*(const v8bf*)(qp + kk * 32 + 16 * g),
                    *(const v8bf*)(qp + kk * 32 + 16 * g + 8));
  }

  v8f o[8];
  float rmax[8], rsum[8];
#pragma unroll
  for (int t = 0; t < 8; ++t) o[t] = zero8();
#pragma unroll
  for (int r = 0; r < 8; ++r) { rmax[r] = -1e30f; rsum[r] = 0.0f; }

  const int nsteps = (qbase >> 5) + 4; // causal bound for this 128-row chunk
  stage(0, 0);
  for (int jt = 0; jt < nsteps; ++jt) {
    const int j0 = jt << 5;
    const int cur = jt & 1;
#if USE_ASYNC_LDS
    wait_async0();
    __syncthreads();
    if (jt + 1 < nsteps) stage(cur ^ 1, j0 + 32);
#else
    __syncthreads();
    if (jt + 1 < nsteps) stage(cur ^ 1, j0 + 32);
    __syncthreads();
#endif

    // S (16x32) = Q . K^T, two 16x16 N-tiles, K-depth 128.
    v8f s0 = zero8(), s1 = zero8();
#pragma unroll
    for (int kk = 0; kk < 4; ++kk) {
      v16bf b0 = *(const v16bf*)&Ks[cur][lm][kk * 32 + 16 * g];
      v16bf b1 = *(const v16bf*)&Ks[cur][16 + lm][kk * 32 + 16 * g];
      s0 = __builtin_amdgcn_wmma_f32_16x16x32_bf16(false, aq[kk], false, b0,
                                                   (short)0, s0, false, false);
      s1 = __builtin_amdgcn_wmma_f32_16x16x32_bf16(false, aq[kk], false, b1,
                                                   (short)0, s1, false, false);
    }

    // Causal mask + online softmax; emit P (bf16) into per-wave LDS.
#pragma unroll
    for (int r = 0; r < 8; ++r) {
      const int qrow = qbase + w * 16 + g * 8 + r;
      const float v0 = (j0 + lm <= qrow) ? s0[r] : -1e30f;
      const float v1 = (j0 + 16 + lm <= qrow) ? s1[r] : -1e30f;
      const float cm = xlane16_max(fmaxf(v0, v1));
      const float nm = fmaxf(rmax[r], cm);
      const float alpha = __expf(rmax[r] - nm);
      rmax[r] = nm;
      const float e0 = (v0 > -1e29f) ? __expf(v0 - nm) : 0.0f;
      const float e1 = (v1 > -1e29f) ? __expf(v1 - nm) : 0.0f;
      const float ps = xlane16_sum(e0 + e1);
      rsum[r] = rsum[r] * alpha + ps;
#pragma unroll
      for (int t = 0; t < 8; ++t) o[t][r] *= alpha;
      Ps[w][g * 8 + r][lm] = f2bf(e0);
      Ps[w][g * 8 + r][16 + lm] = f2bf(e1);
    }
    asm volatile("s_wait_dscnt 0" ::: "memory"); // order same-wave LDS RAW

    // O (16x128) += P (16x32) @ V (32x128)
    v16bf pf = cat8(*(const v8bf*)&Ps[w][lm][16 * g],
                    *(const v8bf*)&Ps[w][lm][16 * g + 8]);
#pragma unroll
    for (int t = 0; t < 8; ++t) {
      v16bf vf = *(const v16bf*)&Vt[cur][t * 16 + lm][16 * g];
      o[t] = __builtin_amdgcn_wmma_f32_16x16x32_bf16(false, pf, false, vf,
                                                     (short)0, o[t], false, false);
    }
#if !USE_ASYNC_LDS
    __syncthreads();
#endif
  }

  // Normalize and write O tile (bf16) back to [MTOT, D_MODEL] layout.
#pragma unroll
  for (int r = 0; r < 8; ++r) {
    const int qrow = qbase + w * 16 + g * 8 + r;
    const float inv = 1.0f / rsum[r];
#pragma unroll
    for (int t = 0; t < 8; ++t)
      O[(rowOff + qrow) * D_MODEL + colOff + t * 16 + lm] = f2bf(o[t][r] * inv);
  }
}

// ---------------------------------------------------------------------------
// Host-side orchestration
// ---------------------------------------------------------------------------
extern "C" void kernel_launch(void* const* d_in, const int* in_sizes, int n_in,
                              void* d_out, int out_size, void* d_ws,
                              size_t ws_size, hipStream_t stream) {
  (void)in_sizes; (void)n_in; (void)out_size; (void)ws_size;
  const float* x = (const float*)d_in[0];
  const float* wq = (const float*)d_in[1];
  const float* wk = (const float*)d_in[2];
  const float* wv = (const float*)d_in[3];
  const float* wo = (const float*)d_in[4];

  char* ws = (char*)d_ws;
  __bf16* xb  = (__bf16*)(ws + 0);
  __bf16* wqb = (__bf16*)(ws + 16777216);
  __bf16* wkb = (__bf16*)(ws + 25165824);
  __bf16* wvb = (__bf16*)(ws + 33554432);
  __bf16* wob = (__bf16*)(ws + 41943040);
  __bf16* Qb  = (__bf16*)(ws + 50331648);
  __bf16* Kb  = (__bf16*)(ws + 67108864);
  __bf16* Vb  = (__bf16*)(ws + 83886080);
  __bf16* Ob  = (__bf16*)(ws + 100663296);
  __bf16* Vtg = (__bf16*)(ws + 117440512); // [32][128][2048], ends at 128MB

  const int nx = MTOT * D_MODEL;     // 8,388,608
  const int nw = D_MODEL * D_MODEL;  // 4,194,304
  cvt_f32_bf16<<<nx / 1024, 256, 0, stream>>>(x, xb, nx / 4);
  cvt_f32_bf16<<<nw / 1024, 256, 0, stream>>>(wq, wqb, nw / 4);
  cvt_f32_bf16<<<nw / 1024, 256, 0, stream>>>(wk, wkb, nw / 4);
  cvt_f32_bf16<<<nw / 1024, 256, 0, stream>>>(wv, wvb, nw / 4);
  cvt_f32_bf16<<<nw / 1024, 256, 0, stream>>>(wo, wob, nw / 4);

  const float sm_scale = 0.08838834764831845f; // 1/sqrt(HEAD_DIM)
  dim3 gg(D_MODEL / 256, MTOT / 128); // (8, 32)
  gemm_bf16t<false><<<gg, 256, 0, stream>>>(xb, wqb, Qb, MTOT, D_MODEL, D_MODEL, sm_scale);
  gemm_bf16t<false><<<gg, 256, 0, stream>>>(xb, wkb, Kb, MTOT, D_MODEL, D_MODEL, 1.0f);
  gemm_bf16t<false><<<gg, 256, 0, stream>>>(xb, wvb, Vb, MTOT, D_MODEL, D_MODEL, 1.0f);

  dim3 gt(SEQ / 64, HEAD_DIM / 64, BATCH * NUM_HEAD); // (32, 2, 32)
  transpose_v<<<gt, 256, 0, stream>>>(Vb, Vtg);

  dim3 ga(SEQ / 128, BATCH * NUM_HEAD); // (16, 32)
  attn_fwd<<<ga, 256, 0, stream>>>(Qb, Kb, Vtg, Ob);

  gemm_bf16t<true><<<gg, 256, 0, stream>>>(Ob, wob, d_out, MTOT, D_MODEL, D_MODEL, 1.0f);
}
